// RoPEMultiheadAttention_2379411882457
// MI455X (gfx1250) — matrix operand order repaired and checked by reading, not compiled, not run on hardware
//
#include <hip/hip_runtime.h>
#include <hip/hip_bf16.h>

// ---------------------------------------------------------------------------
// RoPE multi-head attention for MI455X (gfx1250), bf16 WMMA pipeline.
// B=2, S=2048, E=1024, H=16, HD=64.
// Round 3: async global->LDS staging via GLOBAL_LOAD_ASYNC_TO_LDS_B128 with
// the correctly-typed builtin signature (v4i AS1*, v4i AS3*, imm, imm).
// ---------------------------------------------------------------------------

typedef __bf16 v16bf __attribute__((ext_vector_type(16)));
typedef __bf16 v8bf  __attribute__((ext_vector_type(8)));
typedef float  v8f   __attribute__((ext_vector_type(8)));
typedef int    v4i   __attribute__((ext_vector_type(4)));

typedef __attribute__((address_space(1))) v4i* gptr_v4i;
typedef __attribute__((address_space(3))) v4i* sptr_v4i;

constexpr int BATCH = 2;
constexpr int SEQ   = 2048;
constexpr int EMB   = 1024;
constexpr int NH    = 16;
constexpr int HD    = 64;
constexpr int MROWS = BATCH * SEQ;   // 4096
constexpr float ATTN_SCALE = 0.125f; // HD^-0.5

union V16U { v16bf v; v8bf h[2]; };

#if defined(__HIP_DEVICE_COMPILE__) && __has_builtin(__builtin_amdgcn_global_load_async_to_lds_b128)
#define HAVE_ASYNC_LDS 1
#else
#define HAVE_ASYNC_LDS 0
#endif

// 16-byte global -> LDS copy (async DMA path, ASYNCcnt-tracked).
__device__ inline void cp16_g2s(__bf16* lds, const __bf16* g) {
#if HAVE_ASYNC_LDS
    __builtin_amdgcn_global_load_async_to_lds_b128(
        (gptr_v4i)g, (sptr_v4i)lds, /*imm offset=*/0, /*cpol=*/0);
#else
    *(v8bf*)lds = *(const v8bf*)g;
#endif
}

// Drain ASYNCcnt (producer side) then workgroup barrier.
__device__ inline void fill_fence_barrier() {
#if HAVE_ASYNC_LDS
#if __has_builtin(__builtin_amdgcn_s_wait_asynccnt)
    __builtin_amdgcn_s_wait_asynccnt(0);
#else
    asm volatile("s_wait_asynccnt 0" ::: "memory");
#endif
#endif
    __syncthreads();
}

__device__ inline v8f wmma_bf16(v16bf a, v16bf b, v8f c) {
    // D = A(16x32 bf16) x B(32x16 bf16) + C(16x16 f32)
    return __builtin_amdgcn_wmma_f32_16x16x32_bf16(
        /*neg_a=*/false, a, /*neg_b=*/false, b,
        /*c_mod=*/(short)0, c, /*reuse_a=*/false, /*reuse_b=*/false);
}

// ---------------------------------------------------------------------------
// fp32 -> bf16 conversion
// ---------------------------------------------------------------------------
__global__ __launch_bounds__(256)
void cvt_f32_bf16_kernel(const float* __restrict__ src, __bf16* __restrict__ dst, int n) {
    int i = blockIdx.x * blockDim.x + threadIdx.x;
    if (i < n) dst[i] = (__bf16)src[i];
}

// ---------------------------------------------------------------------------
// Tiled bf16 WMMA GEMM: OUT[m,n] = sum_k A[m,k] * W[n,k] + bias[n]
// Block tile 128x64, 256 threads = 8 waves, each wave 32x32 (2x2 WMMA frags).
// mode 0: Q-proj  (+RoPE, bf16 out at [b,h,s,d])
// mode 1: K-proj  (+RoPE, bf16 out at [b,h,s,d])
// mode 2: V-proj  (bf16 out transposed at [b,h,d,s])
// mode 3: O-proj  (fp32 out at [m,n])
// ---------------------------------------------------------------------------
__global__ __launch_bounds__(256)
void gemm_bf16_kernel(const __bf16* __restrict__ A, const __bf16* __restrict__ W,
                      const float* __restrict__ bias, void* __restrict__ outp,
                      int mode)
{
    __shared__ __align__(16) __bf16 Xs[128][48];
    __shared__ __align__(16) __bf16 Ws[64][48];

    const int tid  = threadIdx.x;
    const int lane = tid & 31;
    const int lh   = lane >> 4;      // lane half (0: lanes 0-15, 1: lanes 16-31)
    const int lr   = lane & 15;
    const int wave = tid >> 5;
    const int wm   = wave & 3;       // 4 row-groups of 32
    const int wn   = wave >> 2;      // 2 col-groups of 32
    const int mtile = blockIdx.y * 128;
    const int ntile = blockIdx.x * 64;

    const v8f vzero = {0.f,0.f,0.f,0.f,0.f,0.f,0.f,0.f};
    v8f acc[2][2];
    #pragma unroll
    for (int i = 0; i < 2; i++)
        #pragma unroll
        for (int j = 0; j < 2; j++) acc[i][j] = vzero;

    for (int kk = 0; kk < EMB; kk += 32) {
        // --- cooperative LDS fill (async global->LDS DMA) ---
        {
            int r  = tid >> 1;                  // 0..127
            int cs = (tid & 1) * 16;            // 0 or 16
            const __bf16* gp = A + (size_t)(mtile + r) * EMB + kk + cs;
            if (kk + 32 < EMB) __builtin_prefetch(gp + 32, 0, 0);
            cp16_g2s(&Xs[r][cs],     gp);
            cp16_g2s(&Xs[r][cs + 8], gp + 8);
        }
        {
            int r  = tid >> 2;                  // 0..63
            int cs = (tid & 3) * 8;             // 0,8,16,24
            const __bf16* gp = W + (size_t)(ntile + r) * EMB + kk + cs;
            cp16_g2s(&Ws[r][cs], gp);
        }
        fill_fence_barrier();

        // --- fragment loads per documented wave32 layouts ---
        V16U af[2], bf[2];
        #pragma unroll
        for (int i = 0; i < 2; i++) {
            int row = wm * 32 + i * 16 + lr;
            af[i].h[0] = *(const v8bf*)&Xs[row][lh * 8];        // K 0-7 / 8-15
            af[i].h[1] = *(const v8bf*)&Xs[row][16 + lh * 8];   // K 16-23 / 24-31
        }
        #pragma unroll
        for (int j = 0; j < 2; j++) {
            int n = wn * 32 + j * 16 + lr;
            bf[j].h[0] = *(const v8bf*)&Ws[n][lh * 8];          // K 0-7 / 8-15
            bf[j].h[1] = *(const v8bf*)&Ws[n][16 + lh * 8];
        }
        #pragma unroll
        for (int i = 0; i < 2; i++)
            #pragma unroll
            for (int j = 0; j < 2; j++)
                acc[i][j] = wmma_bf16(af[i].v, bf[j].v, acc[i][j]);
        __syncthreads();
    }

    // --- epilogue: bias (+RoPE) and scatter per mode ---
    #pragma unroll
    for (int i = 0; i < 2; i++) {
        #pragma unroll
        for (int j = 0; j < 2; j++) {
            int col = ntile + wn * 32 + j * 16 + lr;
            float bval = bias[col];
            #pragma unroll
            for (int r = 0; r < 8; r++) {
                int row = mtile + wm * 32 + i * 16 + r + lh * 8;
                float v = acc[i][j][r] + bval;
                if (mode == 3) {
                    ((float*)outp)[(size_t)row * EMB + col] = v;
                } else {
                    int bb = row / SEQ, s = row % SEQ;
                    int hh = col / HD,  d = col % HD;
                    if (mode <= 1) {
                        // RoPE (reference formula: o2 = e2*sin + o*cos)
                        int   i2 = d >> 1;
                        float theta = (float)s *
                            __powf(10000.0f, -((float)(2 * i2)) * (1.0f / (float)HD));
                        float sn, cs;
                        __sincosf(theta, &sn, &cs);
                        float other = __shfl_xor(v, 1, 32);  // pair partner lane
                        float e  = (d & 1) ? other : v;
                        float o  = (d & 1) ? v     : other;
                        float e2 = e * cs - o * sn;
                        v = (d & 1) ? (e2 * sn + o * cs) : e2;
                    }
                    __bf16* op = (__bf16*)outp;
                    size_t idx = (mode == 2)
                        ? ((size_t)((bb * NH + hh) * HD + d) * SEQ + s)   // V^T
                        : ((size_t)((bb * NH + hh) * SEQ + s) * HD + d);  // Q/K
                    op[idx] = (__bf16)v;
                }
            }
        }
    }
}

// ---------------------------------------------------------------------------
// Flash attention: grid (S/64, B*H), 128 threads = 4 waves x 16 query rows.
// Q frags resident in VGPRs; K and V^T tiles async-staged in LDS; online
// softmax; P converted C-layout -> A-layout through per-wave LDS scratch.
// ---------------------------------------------------------------------------
__global__ __launch_bounds__(128)
void flash_attn_kernel(const __bf16* __restrict__ Q, const __bf16* __restrict__ K,
                       const __bf16* __restrict__ Vt, __bf16* __restrict__ Ao)
{
    __shared__ __align__(16) __bf16 Ks[64][72];
    __shared__ __align__(16) __bf16 Vs[64][72];
    __shared__ __align__(16) __bf16 Ps[4][16][72];

    const int tid  = threadIdx.x;
    const int lane = tid & 31;
    const int lh   = lane >> 4;
    const int lr   = lane & 15;
    const int wave = tid >> 5;
    const int bh   = blockIdx.y;
    const int b    = bh / NH, h = bh % NH;
    const int qbase = blockIdx.x * 64 + wave * 16;

    // Q fragments (16 rows x 64 head-dim = two 16x32 A frags), from global.
    V16U Aq[2];
    {
        const __bf16* qp = Q + ((size_t)bh * SEQ + qbase + lr) * HD;
        #pragma unroll
        for (int f = 0; f < 2; f++) {
            Aq[f].h[0] = *(const v8bf*)(qp + f * 32 + lh * 8);
            Aq[f].h[1] = *(const v8bf*)(qp + f * 32 + 16 + lh * 8);
        }
    }

    const v8f vzero = {0.f,0.f,0.f,0.f,0.f,0.f,0.f,0.f};
    v8f Of[4];
    float m_run[8], l_run[8];
    #pragma unroll
    for (int d = 0; d < 4; d++) Of[d] = vzero;
    #pragma unroll
    for (int r = 0; r < 8; r++) { m_run[r] = -3.0e38f; l_run[r] = 0.f; }

    for (int kt = 0; kt < SEQ; kt += 64) {
        // --- cooperative LDS fill: K tile [key][d], V^T tile [d][key] ---
        {
            int r  = tid >> 1;                 // 0..63
            int cs = (tid & 1) * 32;
            const __bf16* kp = K  + ((size_t)bh * SEQ + kt + r) * HD + cs;
            const __bf16* vp = Vt + ((size_t)bh * HD  + r) * SEQ + kt + cs;
            #pragma unroll
            for (int u = 0; u < 4; u++) {
                cp16_g2s(&Ks[r][cs + u * 8], kp + u * 8);
                cp16_g2s(&Vs[r][cs + u * 8], vp + u * 8);
            }
        }
        fill_fence_barrier();

        // --- S = Q K^T for 64 keys (4 N-frags, K-chain of 2 WMMAs each) ---
        v8f Sf[4];
        #pragma unroll
        for (int nf = 0; nf < 4; nf++) {
            int key = nf * 16 + lr;
            V16U b0, b1;
            b0.h[0] = *(const v8bf*)&Ks[key][lh * 8];
            b0.h[1] = *(const v8bf*)&Ks[key][16 + lh * 8];
            b1.h[0] = *(const v8bf*)&Ks[key][32 + lh * 8];
            b1.h[1] = *(const v8bf*)&Ks[key][32 + 16 + lh * 8];
            v8f s = vzero;
            s = wmma_bf16(Aq[0].v, b0.v, s);
            s = wmma_bf16(Aq[1].v, b1.v, s);
            Sf[nf] = s;
        }

        // --- online softmax (rows live in 16-lane halves of the C layout) ---
        #pragma unroll
        for (int nf = 0; nf < 4; nf++)
            #pragma unroll
            for (int r = 0; r < 8; r++) Sf[nf][r] *= ATTN_SCALE;

        float mt[8], ls[8], alpha[8];
        #pragma unroll
        for (int r = 0; r < 8; r++) {
            mt[r] = fmaxf(fmaxf(Sf[0][r], Sf[1][r]), fmaxf(Sf[2][r], Sf[3][r]));
            #pragma unroll
            for (int off = 8; off > 0; off >>= 1)
                mt[r] = fmaxf(mt[r], __shfl_xor(mt[r], off, 16));
            float mn = fmaxf(m_run[r], mt[r]);
            alpha[r] = __expf(m_run[r] - mn);
            m_run[r] = mn;
            ls[r] = 0.f;
        }
        #pragma unroll
        for (int nf = 0; nf < 4; nf++)
            #pragma unroll
            for (int r = 0; r < 8; r++) {
                float p = __expf(Sf[nf][r] - m_run[r]);
                Sf[nf][r] = p;
                ls[r] += p;
            }
        #pragma unroll
        for (int r = 0; r < 8; r++) {
            #pragma unroll
            for (int off = 8; off > 0; off >>= 1)
                ls[r] += __shfl_xor(ls[r], off, 16);
            l_run[r] = l_run[r] * alpha[r] + ls[r];
        }
        #pragma unroll
        for (int d = 0; d < 4; d++)
            #pragma unroll
            for (int r = 0; r < 8; r++) Of[d][r] *= alpha[r];

        // --- C-layout P -> LDS -> A-layout frags (same-wave DS ordering) ---
        #pragma unroll
        for (int nf = 0; nf < 4; nf++)
            #pragma unroll
            for (int r = 0; r < 8; r++)
                Ps[wave][r + lh * 8][nf * 16 + lr] = (__bf16)Sf[nf][r];

        V16U Ap[2];
        #pragma unroll
        for (int f = 0; f < 2; f++) {
            Ap[f].h[0] = *(const v8bf*)&Ps[wave][lr][f * 32 + lh * 8];
            Ap[f].h[1] = *(const v8bf*)&Ps[wave][lr][f * 32 + 16 + lh * 8];
        }

        // --- O += P V (B frags contiguous thanks to V^T layout) ---
        #pragma unroll
        for (int df = 0; df < 4; df++) {
            int d = df * 16 + lr;
            V16U b0, b1;
            b0.h[0] = *(const v8bf*)&Vs[d][lh * 8];
            b0.h[1] = *(const v8bf*)&Vs[d][16 + lh * 8];
            b1.h[0] = *(const v8bf*)&Vs[d][32 + lh * 8];
            b1.h[1] = *(const v8bf*)&Vs[d][32 + 16 + lh * 8];
            Of[df] = wmma_bf16(Ap[0].v, b0.v, Of[df]);
            Of[df] = wmma_bf16(Ap[1].v, b1.v, Of[df]);
        }
        __syncthreads();
    }

    // --- normalize and write bf16 [b, s, e] for the O projection ---
    #pragma unroll
    for (int df = 0; df < 4; df++) {
        int e = h * HD + df * 16 + lr;
        #pragma unroll
        for (int r = 0; r < 8; r++) {
            int s = qbase + r + lh * 8;
            float v = Of[df][r] / l_run[r];
            Ao[((size_t)b * SEQ + s) * EMB + e] = (__bf16)v;
        }
    }
}

// ---------------------------------------------------------------------------
// Host launcher
// ---------------------------------------------------------------------------
extern "C" void kernel_launch(void* const* d_in, const int* in_sizes, int n_in,
                              void* d_out, int out_size, void* d_ws, size_t ws_size,
                              hipStream_t stream)
{
    (void)in_sizes; (void)n_in; (void)out_size; (void)ws_size;
    const float* query = (const float*)d_in[0];
    const float* Wq = (const float*)d_in[1];
    const float* bq = (const float*)d_in[2];
    const float* Wk = (const float*)d_in[3];
    const float* bk = (const float*)d_in[4];
    const float* Wv = (const float*)d_in[5];
    const float* bv = (const float*)d_in[6];
    const float* Wo = (const float*)d_in[7];
    const float* bo = (const float*)d_in[8];
    float* out = (float*)d_out;

    char* ws = (char*)d_ws;
    size_t off = 0;
    auto alloc = [&](size_t bytes) -> char* {
        char* p = ws + off;
        off += (bytes + 255) & ~(size_t)255;
        return p;
    };
    const size_t NME  = (size_t)MROWS * EMB;       // 4096 x 1024
    const size_t NEE  = (size_t)EMB * EMB;         // 1024 x 1024
    const size_t NBHS = (size_t)BATCH * NH * SEQ * HD;

    __bf16* Xb  = (__bf16*)alloc(NME  * 2);
    __bf16* Wqb = (__bf16*)alloc(NEE  * 2);
    __bf16* Wkb = (__bf16*)alloc(NEE  * 2);
    __bf16* Wvb = (__bf16*)alloc(NEE  * 2);
    __bf16* Wob = (__bf16*)alloc(NEE  * 2);
    __bf16* Qb  = (__bf16*)alloc(NBHS * 2);
    __bf16* Kb  = (__bf16*)alloc(NBHS * 2);
    __bf16* Vtb = (__bf16*)alloc(NBHS * 2);
    __bf16* Ab  = (__bf16*)alloc(NME  * 2);

    // Stage 1: fp32 -> bf16 conversions
    cvt_f32_bf16_kernel<<<(int)((NME + 255) / 256), 256, 0, stream>>>(query, Xb, (int)NME);
    cvt_f32_bf16_kernel<<<(int)((NEE + 255) / 256), 256, 0, stream>>>(Wq, Wqb, (int)NEE);
    cvt_f32_bf16_kernel<<<(int)((NEE + 255) / 256), 256, 0, stream>>>(Wk, Wkb, (int)NEE);
    cvt_f32_bf16_kernel<<<(int)((NEE + 255) / 256), 256, 0, stream>>>(Wv, Wvb, (int)NEE);
    cvt_f32_bf16_kernel<<<(int)((NEE + 255) / 256), 256, 0, stream>>>(Wo, Wob, (int)NEE);

    // Stage 2: QKV projections (Q/K get RoPE; V written transposed)
    dim3 ggrid(EMB / 64, MROWS / 128);
    gemm_bf16_kernel<<<ggrid, 256, 0, stream>>>(Xb, Wqb, bq, (void*)Qb,  0);
    gemm_bf16_kernel<<<ggrid, 256, 0, stream>>>(Xb, Wkb, bk, (void*)Kb,  1);
    gemm_bf16_kernel<<<ggrid, 256, 0, stream>>>(Xb, Wvb, bv, (void*)Vtb, 2);

    // Stage 3: flash attention
    dim3 agrid(SEQ / 64, BATCH * NH);
    flash_attn_kernel<<<agrid, 128, 0, stream>>>(Qb, Kb, Vtb, Ab);

    // Stage 4: output projection (fp32 out)
    gemm_bf16_kernel<<<ggrid, 256, 0, stream>>>(Ab, Wob, bo, (void*)out, 3);
}
